// GraphSAGE_79920751444224
// MI455X (gfx1250) — matrix operand order repaired and checked by reading, not compiled, not run on hardware
//
#include <hip/hip_runtime.h>

// CDNA5 (gfx1250) wave32 WMMA types
typedef __attribute__((ext_vector_type(16))) _Float16 v16h;
typedef __attribute__((ext_vector_type(8)))  float    v8f;

#define WAVE 32

__device__ __forceinline__ v8f wmma_f16(v16h a, v16h b, v8f c) {
  // D = A(16x32 f16) x B(32x16 f16) + C(16x16 f32)
  return __builtin_amdgcn_wmma_f32_16x16x32_f16(
      /*neg_a=*/false, a, /*neg_b=*/false, b,
      /*c_mod=*/(short)0, c, /*reuse_a=*/false, /*reuse_b=*/false);
}

// Pack two float4 (8 consecutive f32) into half of a v16h, scaled.
__device__ __forceinline__ void pack8(v16h& a, int off, float4 u, float4 v, float s) {
  a[off + 0] = (_Float16)(u.x * s); a[off + 1] = (_Float16)(u.y * s);
  a[off + 2] = (_Float16)(u.z * s); a[off + 3] = (_Float16)(u.w * s);
  a[off + 4] = (_Float16)(v.x * s); a[off + 5] = (_Float16)(v.y * s);
  a[off + 6] = (_Float16)(v.z * s); a[off + 7] = (_Float16)(v.w * s);
}

__global__ void zero_kernel(float4* __restrict__ p, long n4) {
  long i = (long)blockIdx.x * blockDim.x + threadIdx.x;
  long stride = (long)gridDim.x * blockDim.x;
  float4 z; z.x = z.y = z.z = z.w = 0.f;
  for (; i < n4; i += stride) p[i] = z;
}

// Fused: gather x[src] -> scale by (1+coef*w) -> [16x64]@[64x64] WMMA GEMM
// -> +bias, relu -> atomic-max scatter into agg[dst].
// agg has N+1 rows, pre-zeroed. Self-loops and tail rows map to dummy row N
// (same trick as the reference's dummy segment) -> output loop is branchless.
// relu output >= 0 so uint-bitpattern atomicMax == float max, and zero-init
// reproduces the reference's isfinite->0 handling of empty segments.
__global__ void __launch_bounds__(256, 1) edge_pool_max_kernel(
    const float* __restrict__ xin,   // [N,64]
    const int*   __restrict__ esrc,  // [E]
    const int*   __restrict__ edst,  // [E]
    const float* __restrict__ ew,    // [E]
    const float* __restrict__ coefp, // [1]
    const float* __restrict__ pw,    // [64,64] (out_ch, in_ch)
    const float* __restrict__ pb,    // [64]
    float*       __restrict__ agg,   // [N+1,64], pre-zeroed
    int E2, int E, int N)
{
  const int lane = threadIdx.x & (WAVE - 1);
  const int wid  = blockIdx.x * (blockDim.x / WAVE) + (threadIdx.x / WAVE);
  const int nw   = gridDim.x * (blockDim.x / WAVE);
  const int col  = lane & 15;
  const int hi   = lane >> 4;         // which 16-lane half
  const float coef = coefp[0];

  // Preload B = pw^T fragments: B[k][n] = pw[n][k].
  // 32x16 f16 B layout: lanes 0-15 hold K=0..7,16..23, lanes 16-31 K=8..15,24..31;
  // element t<8 = B[kbase+hi*8+t][col], t>=8 = B[kbase+16+hi*8+t-8][col] -> 4x b128.
  v16h bfrag[4][2];
  #pragma unroll
  for (int ct = 0; ct < 4; ++ct) {
    #pragma unroll
    for (int kc = 0; kc < 2; ++kc) {
      const float4* w4 = (const float4*)(pw + (ct * 16 + col) * 64 + kc * 32 + hi * 8);
      float4 q0 = w4[0], q1 = w4[1], q2 = w4[4], q3 = w4[5];
      v16h b;
      pack8(b, 0, q0, q1, 1.f);
      pack8(b, 8, q2, q3, 1.f);
      bfrag[ct][kc] = b;
    }
  }

  // Bias is tile-invariant: hoist out of the edge loop.
  float bias[4];
  #pragma unroll
  for (int ct = 0; ct < 4; ++ct) bias[ct] = pb[ct * 16 + col];

  const int ntiles = (E2 + 15) >> 4;
  for (int tile = wid; tile < ntiles; tile += nw) {
    // Row 'col' of this 16-edge tile (lanes r and r+16 hold the same edge).
    int e = tile * 16 + col;
    int s = 0, d = 0;
    float scale = 0.f;
    if (e < E2) {
      if (e < E) { s = esrc[e];     d = edst[e];     scale = 1.f + coef * ew[e];     }
      else       { s = edst[e - E]; d = esrc[e - E]; scale = 1.f + coef * ew[e - E]; }
    }
    // Self-loop / tail rows -> dummy row N (written, never read).
    int pk = (s != d) ? d : N;

    // Per-output-row destination, resolved early with constant-lane readlanes
    // and one cndmask each (pk replicated across the two 16-lane halves).
    // D layout: lane l -> n = l&15, VGPR v -> row = v + (l>>4)*8.
    int dr[8];
    #pragma unroll
    for (int v = 0; v < 8; ++v) {
      int lo  = __builtin_amdgcn_readlane(pk, v);
      int hi8 = __builtin_amdgcn_readlane(pk, v + 8);
      dr[v] = hi ? hi8 : lo;
    }

    // A fragments (16x32 f16, §7.12.2). Element pairs are contiguous:
    // elems 0..7  = feat[kc*32 +      hi*8 + 0..7]
    // elems 8..15 = feat[kc*32 + 16 + hi*8 + 0..7]   -> 4x b128 per fragment
    const float4* xr4 = (const float4*)(xin + (long)s * 64);
    v16h afrag[2];
    #pragma unroll
    for (int kc = 0; kc < 2; ++kc) {
      float4 p0 = xr4[kc * 8 + hi * 2 + 0];
      float4 p1 = xr4[kc * 8 + hi * 2 + 1];
      float4 p2 = xr4[kc * 8 + 4 + hi * 2 + 0];
      float4 p3 = xr4[kc * 8 + 4 + hi * 2 + 1];
      v16h a;
      pack8(a, 0, p0, p1, scale);
      pack8(a, 8, p2, p3, scale);
      afrag[kc] = a;
    }

    // All four 16-col output tiles.
    v8f acc[4];
    #pragma unroll
    for (int ct = 0; ct < 4; ++ct) {
      v8f c = {};
      c = wmma_f16(afrag[0], bfrag[ct][0], c);
      c = wmma_f16(afrag[1], bfrag[ct][1], c);
      acc[ct] = c;
    }

    // Branchless: one address per row; 4 atomics share the base address via
    // immediate offsets (single s_clause). relu via v_max_num_f32.
    #pragma unroll
    for (int v = 0; v < 8; ++v) {
      unsigned int* rowp = (unsigned int*)(agg + (long)dr[v] * 64 + col);
      #pragma unroll
      for (int ct = 0; ct < 4; ++ct) {
        float val = fmaxf(acc[ct][v] + bias[ct], 0.f);
        atomicMax(rowp + ct * 16, __float_as_uint(val));
      }
    }
  }
}

// h = relu([xin | agg] @ fw^T + fb), fw: [64,128]
__global__ void __launch_bounds__(256, 1) fin_kernel(
    const float* __restrict__ xin,  // [N,64]
    const float* __restrict__ agg,  // [N(+1),64] (row N unused here)
    const float* __restrict__ fw,   // [64,128]
    const float* __restrict__ fb,   // [64]
    float*       __restrict__ hout, // [N,64]
    int N)
{
  const int lane = threadIdx.x & (WAVE - 1);
  const int wid  = blockIdx.x * (blockDim.x / WAVE) + (threadIdx.x / WAVE);
  const int nw   = gridDim.x * (blockDim.x / WAVE);
  const int col  = lane & 15;
  const int hi   = lane >> 4;
  const int ntiles = (N + 15) >> 4;

  float bias[4];
  #pragma unroll
  for (int ct = 0; ct < 4; ++ct) bias[ct] = fb[ct * 16 + col];

  for (int tile = wid; tile < ntiles; tile += nw) {
    int arow = tile * 16 + col;
    int ld   = (arow < N) ? arow : 0;
    const float4* xr4 = (const float4*)(xin + (long)ld * 64);
    const float4* ar4 = (const float4*)(agg + (long)ld * 64);

    // A = [x | agg], 16x128: kc 0,1 -> x cols 0..63; kc 2,3 -> agg cols 0..63.
    v16h afrag[4];
    #pragma unroll
    for (int kc = 0; kc < 4; ++kc) {
      const float4* src4 = (kc < 2) ? xr4 : ar4;
      int kb = (kc & 1) * 8;                      // float4 index of 32-col chunk
      float4 p0 = src4[kb + hi * 2 + 0];
      float4 p1 = src4[kb + hi * 2 + 1];
      float4 p2 = src4[kb + 4 + hi * 2 + 0];
      float4 p3 = src4[kb + 4 + hi * 2 + 1];
      v16h a;
      pack8(a, 0, p0, p1, 1.f);
      pack8(a, 8, p2, p3, 1.f);
      afrag[kc] = a;
    }

    #pragma unroll
    for (int ct = 0; ct < 4; ++ct) {
      v8f acc = {};
      #pragma unroll
      for (int kc = 0; kc < 4; ++kc) {
        const float4* w4 = (const float4*)(fw + (ct * 16 + col) * 128 + kc * 32 + hi * 8);
        float4 q0 = w4[0], q1 = w4[1], q2 = w4[4], q3 = w4[5];
        v16h b;
        pack8(b, 0, q0, q1, 1.f);
        pack8(b, 8, q2, q3, 1.f);
        acc = wmma_f16(afrag[kc], b, acc);
      }
      #pragma unroll
      for (int v = 0; v < 8; ++v) {
        int orow = tile * 16 + v + hi * 8;
        if (orow < N) {
          hout[(long)orow * 64 + ct * 16 + col] = fmaxf(acc[v] + bias[ct], 0.f);
        }
      }
    }
  }
}

// Per prediction edge: emb = [h[a], h[b]] (128), two 128-d dot products.
__global__ void __launch_bounds__(256) head_kernel(
    const float* __restrict__ h,      // [N,64]
    const int*   __restrict__ pa,     // [P]
    const int*   __restrict__ pbidx,  // [P]
    const float* __restrict__ ewp_w,  // [128]
    const float* __restrict__ ewp_b,  // [1]
    const float* __restrict__ ep_w,   // [128]
    const float* __restrict__ ep_b,   // [1]
    float*       __restrict__ out,    // [2P]: [0,P)=relu weights, [P,2P)=predictor
    int P)
{
  __shared__ float sw[256];
  int t = threadIdx.x;
  if (t < 128)      sw[t] = ewp_w[t];
  else if (t < 256) sw[t] = ep_w[t - 128];
  __syncthreads();

  int p = blockIdx.x * blockDim.x + threadIdx.x;
  if (p >= P) return;
  const float4* ha4 = (const float4*)(h + (long)pa[p] * 64);
  const float4* hb4 = (const float4*)(h + (long)pbidx[p] * 64);
  const float4* sw4 = (const float4*)sw;

  float acc1 = 0.f, acc2 = 0.f;
  #pragma unroll
  for (int q = 0; q < 16; ++q) {
    float4 va = ha4[q],       vb = hb4[q];
    float4 wa = sw4[q],       wb = sw4[16 + q];   // ewp_w halves
    float4 ua = sw4[32 + q],  ub = sw4[48 + q];   // ep_w halves
    acc1 += va.x * wa.x + va.y * wa.y + va.z * wa.z + va.w * wa.w
          + vb.x * wb.x + vb.y * wb.y + vb.z * wb.z + vb.w * wb.w;
    acc2 += va.x * ua.x + va.y * ua.y + va.z * ua.z + va.w * ua.w
          + vb.x * ub.x + vb.y * ub.y + vb.z * ub.z + vb.w * ub.w;
  }
  out[p]     = fmaxf(acc1 + ewp_b[0], 0.f);
  out[P + p] = acc2 + ep_b[0];
}

extern "C" void kernel_launch(void* const* d_in, const int* in_sizes, int n_in,
                              void* d_out, int out_size, void* d_ws, size_t ws_size,
                              hipStream_t stream) {
  const float* x     = (const float*)d_in[0];
  const int*   pe    = (const int*)  d_in[1];
  const int*   me    = (const int*)  d_in[2];
  const float* mew   = (const float*)d_in[3];
  const float* coef1 = (const float*)d_in[4];
  const float* pw1   = (const float*)d_in[5];
  const float* pb1   = (const float*)d_in[6];
  const float* fw1   = (const float*)d_in[7];
  const float* fb1   = (const float*)d_in[8];
  const float* coef2 = (const float*)d_in[9];
  const float* pw2   = (const float*)d_in[10];
  const float* pb2   = (const float*)d_in[11];
  const float* fw2   = (const float*)d_in[12];
  const float* fb2   = (const float*)d_in[13];
  const float* ewp_w = (const float*)d_in[14];
  const float* ewp_b = (const float*)d_in[15];
  const float* ep_w  = (const float*)d_in[16];
  const float* ep_b  = (const float*)d_in[17];

  const int N  = in_sizes[0] / 64;
  const int E  = in_sizes[3];
  const int P  = in_sizes[1] / 2;
  const int E2 = 2 * E;

  const int* esrc = me;
  const int* edst = me + E;
  const int* pa   = pe;
  const int* pbx  = pe + P;

  // agg buffers carry a dummy row N for self-loop / tail-row sinks.
  float* agg1 = (float*)d_ws;                     // [N+1,64]
  float* h1   = agg1 + (long)(N + 1) * 64;        // [N,64]
  float* agg2 = h1   + (long)N * 64;              // [N+1,64]
  float* h2   = agg2 + (long)(N + 1) * 64;        // [N,64]
  (void)ws_size; (void)n_in; (void)out_size;

  // zero both aggregation buffers (incl. dummy row)
  zero_kernel<<<512, 256, 0, stream>>>((float4*)agg1, (long)(N + 1) * 16);
  zero_kernel<<<512, 256, 0, stream>>>((float4*)agg2, (long)(N + 1) * 16);
  // conv1
  edge_pool_max_kernel<<<1024, 256, 0, stream>>>(
      x, esrc, edst, mew, coef1, pw1, pb1, agg1, E2, E, N);
  fin_kernel<<<512, 256, 0, stream>>>(x, agg1, fw1, fb1, h1, N);
  // conv2
  edge_pool_max_kernel<<<1024, 256, 0, stream>>>(
      h1, esrc, edst, mew, coef2, pw2, pb2, agg2, E2, E, N);
  fin_kernel<<<512, 256, 0, stream>>>(h1, agg2, fw2, fb2, h2, N);
  // heads
  head_kernel<<<(P + 255) / 256, 256, 0, stream>>>(
      h2, pa, pbx, ewp_w, ewp_b, ep_w, ep_b, (float*)d_out, P);
}
